// Decoder_transformer_90383291777612
// MI455X (gfx1250) — matrix operand order repaired and checked by reading, not compiled, not run on hardware
//
#include <hip/hip_runtime.h>
#include <math.h>

// ---------------------------------------------------------------------------
// Types / helpers
// ---------------------------------------------------------------------------
typedef __attribute__((ext_vector_type(16))) __bf16 v16bf;
typedef __attribute__((ext_vector_type(8)))  float  v8f;
typedef unsigned short u16;
typedef unsigned int   u32;
typedef unsigned long long u64;

#define DEV static __device__ __forceinline__

DEV u16 f2bf(float f) {
  union { float f; u32 u; } x; x.f = f;
  u32 r = x.u + 0x7fffu + ((x.u >> 16) & 1u);   // round-to-nearest-even
  return (u16)(r >> 16);
}
DEV float bf2f(u16 h) {
  union { u32 u; float f; } x; x.u = ((u32)h) << 16;
  return x.f;
}

union AFrag { v16bf v; u16 u[16]; uint4 q[2]; };

// sinusoidal positional encoding value PE[t][d], D=256
DEV float pe_val(int t, int d) {
  int i = d >> 1;
  float div = __expf((float)(2 * i) * (-9.210340371976184f / 256.0f));  // ln(10000)
  float ang = (float)t * div;
  return (d & 1) ? __cosf(ang) : __sinf(ang);
}

// ---------------------------------------------------------------------------
// Kernel 1: stable argsort (descending) of 32 caption lengths
// ---------------------------------------------------------------------------
__global__ void prep_kernel(const int* __restrict__ cl,
                            int* __restrict__ order, int* __restrict__ declen) {
  __shared__ int s_ord[32];
  int i = threadIdx.x;
  int li = cl[i];
  int rank = 0;
  for (int j = 0; j < 32; ++j) {
    int lj = cl[j];
    if (lj > li || (lj == li && j < i)) rank++;
  }
  s_ord[rank] = i;
  __syncthreads();
  order[i]  = s_ord[i];
  declen[i] = cl[s_ord[i]] - 1;
}

// ---------------------------------------------------------------------------
// Kernel 2: tail outputs of the returned tuple (caps, dec_len, order)
// ---------------------------------------------------------------------------
__global__ void finalize_kernel(const float* __restrict__ caps,
                                const int* __restrict__ order,
                                const int* __restrict__ declen,
                                float* __restrict__ out) {
  int idx = blockIdx.x * blockDim.x + threadIdx.x;
  const size_t base = (size_t)32 * 511 * 8000;
  if (idx < 16384) {
    int b = idx >> 9, t = idx & 511;
    out[base + idx] = caps[order[b] * 512 + t];
  } else if (idx < 16416) {
    out[base + 16384 + (idx - 16384)] = (float)declen[idx - 16384];
  } else if (idx < 16448) {
    out[base + 16416 + (idx - 16416)] = (float)order[idx - 16416];
  }
}

// ---------------------------------------------------------------------------
// Kernel 3: generic f32 -> bf16 convert (weights)
// ---------------------------------------------------------------------------
__global__ void cvt_bf16(const float* __restrict__ src, u16* __restrict__ dst, int n) {
  int i = blockIdx.x * blockDim.x + threadIdx.x;
  if (i < n) dst[i] = f2bf(src[i]);
}

// ---------------------------------------------------------------------------
// Kernel 4: gather encoder_out by sorted order + convert to bf16
// ---------------------------------------------------------------------------
__global__ void gather_enc(const float* __restrict__ enc, const int* __restrict__ order,
                           u16* __restrict__ dst) {
  const size_t n = (size_t)32 * 256 * 2048;
  for (size_t idx = blockIdx.x * (size_t)blockDim.x + threadIdx.x; idx < n;
       idx += (size_t)gridDim.x * blockDim.x) {
    int b = (int)(idx / (256 * 2048));
    size_t rem = idx - (size_t)b * (256 * 2048);
    dst[idx] = f2bf(enc[(size_t)order[b] * (256 * 2048) + rem]);
  }
}

// ---------------------------------------------------------------------------
// Kernel 5: target embedding  x = (tok/sqrt(V))*w_emb + b_emb + PE
// rows laid out [32, 512] (row 511 duplicates t=510; discarded at the end)
// ---------------------------------------------------------------------------
__global__ void embed_kernel(const float* __restrict__ caps, const int* __restrict__ order,
                             const float* __restrict__ w_emb, const float* __restrict__ b_emb,
                             float* __restrict__ Xf, u16* __restrict__ Xb) {
  int idx = blockIdx.x * blockDim.x + threadIdx.x;
  if (idx >= 16384 * 256) return;
  int d   = idx & 255;
  int row = idx >> 8;
  int b = row >> 9, t0 = row & 511;
  int t = t0 < 511 ? t0 : 510;
  float tok = caps[order[b] * 512 + t] * 0.011180339887498949f;  // 1/sqrt(8000)
  float x = tok * w_emb[d] + b_emb[d] + pe_val(t, d);
  Xf[idx] = x;
  Xb[idx] = f2bf(x);
}

// ---------------------------------------------------------------------------
// Kernel 6: bf16 WMMA GEMM   C[M,N] = A[M,K] @ W[N,K]^T (+bias, epilogue modes)
// 128x64 tile per block, 4 waves, 32x64 strip per wave (2 row-groups).
// Software-pipelined: ping-pong fragment buffers, k unrolled by 64.
// __launch_bounds__(128, 1): ~190 live VGPRs by design; allow the full
// register file (1 wave/EU) so nothing spills to scratch.
// Requires M % 128 == 0, N % 64 == 0, K % 64 == 0 (all shapes here comply).
// mode: 0 = bias->bf16 | 1 = bias+PE(row%256)->bf16 | 2 = bias+ReLU->bf16
//       3 = bias->f32  | 4 = logits (mask + remap into d_out)
// ---------------------------------------------------------------------------
__global__ void __launch_bounds__(128, 1)
gemm_wmma(const u16* __restrict__ A, const u16* __restrict__ W,
          const float* __restrict__ bias,
          float* __restrict__ Cf, u16* __restrict__ Cb,
          int M, int N, int K, int mode,
          const int* __restrict__ declen) {
  const int lane = threadIdx.x & 31;
  const int wave = threadIdx.x >> 5;
  const int g = lane >> 4, ln = lane & 15;
  const int row0 = blockIdx.y * 128 + wave * 32;
  const int col0 = blockIdx.x * 64;

  v8f zero = {0.f, 0.f, 0.f, 0.f, 0.f, 0.f, 0.f, 0.f};
  v8f acc[2][4];
#pragma unroll
  for (int rg = 0; rg < 2; ++rg)
#pragma unroll
    for (int j = 0; j < 4; ++j) acc[rg][j] = zero;

  // A-layout: lane ln holds row ln, k in [8g,8g+8) U [16+8g,16+8g+8)
  const u16* arow[2] = { A + (size_t)(row0 + ln) * K,
                         A + (size_t)(row0 + 16 + ln) * K };
  // B-layout: lane holds column, kk in [8g,8g+8) U [16+8g,16+8g+8)
  const u16* wrow[4];
#pragma unroll
  for (int j = 0; j < 4; ++j) wrow[j] = W + (size_t)(col0 + j * 16 + ln) * K + 8 * g;

  AFrag a0[2], b0[4], a1[2], b1[4];
#pragma unroll
  for (int rg = 0; rg < 2; ++rg) {
    a0[rg].q[0] = *(const uint4*)(arow[rg] + 8 * g);
    a0[rg].q[1] = *(const uint4*)(arow[rg] + 16 + 8 * g);
  }
#pragma unroll
  for (int j = 0; j < 4; ++j) {
    b0[j].q[0] = *(const uint4*)(wrow[j]);
    b0[j].q[1] = *(const uint4*)(wrow[j] + 16);
  }

  for (int k0 = 0; k0 < K; k0 += 64) {
    const int kh = k0 + 32;
    // prefetch odd half while even half computes
#pragma unroll
    for (int rg = 0; rg < 2; ++rg) {
      a1[rg].q[0] = *(const uint4*)(arow[rg] + kh + 8 * g);
      a1[rg].q[1] = *(const uint4*)(arow[rg] + kh + 16 + 8 * g);
    }
#pragma unroll
    for (int j = 0; j < 4; ++j) {
      b1[j].q[0] = *(const uint4*)(wrow[j] + kh);
      b1[j].q[1] = *(const uint4*)(wrow[j] + kh + 16);
    }
#pragma unroll
    for (int rg = 0; rg < 2; ++rg)
#pragma unroll
      for (int j = 0; j < 4; ++j)
        acc[rg][j] = __builtin_amdgcn_wmma_f32_16x16x32_bf16(
            false, a0[rg].v, false, b0[j].v, (short)0, acc[rg][j], false, false);
    // prefetch next even half while odd half computes
    if (k0 + 64 < K) {
      const int kn = k0 + 64;
#pragma unroll
      for (int rg = 0; rg < 2; ++rg) {
        a0[rg].q[0] = *(const uint4*)(arow[rg] + kn + 8 * g);
        a0[rg].q[1] = *(const uint4*)(arow[rg] + kn + 16 + 8 * g);
      }
#pragma unroll
      for (int j = 0; j < 4; ++j) {
        b0[j].q[0] = *(const uint4*)(wrow[j] + kn);
        b0[j].q[1] = *(const uint4*)(wrow[j] + kn + 16);
      }
    }
#pragma unroll
    for (int rg = 0; rg < 2; ++rg)
#pragma unroll
      for (int j = 0; j < 4; ++j)
        acc[rg][j] = __builtin_amdgcn_wmma_f32_16x16x32_bf16(
            false, a1[rg].v, false, b1[j].v, (short)0, acc[rg][j], false, false);
  }

#pragma unroll
  for (int rg = 0; rg < 2; ++rg) {
#pragma unroll
    for (int j = 0; j < 4; ++j) {
      const int col = col0 + j * 16 + ln;
      const float bv = bias ? bias[col] : 0.f;
#pragma unroll
      for (int r = 0; r < 8; ++r) {
        const int row = row0 + rg * 16 + r + 8 * g;  // C-layout: vgpr r -> row r/r+8
        float v = acc[rg][j][r] + bv;
        if (mode == 0) {
          Cb[(size_t)row * N + col] = f2bf(v);
        } else if (mode == 1) {
          v += pe_val(row & 255, col);
          Cb[(size_t)row * N + col] = f2bf(v);
        } else if (mode == 2) {
          v = v > 0.f ? v : 0.f;
          Cb[(size_t)row * N + col] = f2bf(v);
        } else if (mode == 3) {
          Cf[(size_t)row * N + col] = v;
        } else {  // logits: rows are (b,t) with t in [0,512); skip padded t==511
          int b = row >> 9, t = row & 511;
          if (t < 511) {
            float o = (t < declen[b]) ? v : -1.0f;
            Cf[((size_t)(b * 511 + t)) * 8000 + col] = o;
          }
        }
      }
    }
  }
}

// ---------------------------------------------------------------------------
// Kernel 7: flash attention (bf16 WMMA QK^T and PV, f32 online softmax)
// one wave per (b, h, 16-query tile); keys processed in blocks of 32.
// V tiles staged LDS-side via async global->LDS copies (ASYNCcnt).
// ---------------------------------------------------------------------------
__global__ void attn_wmma(const u16* __restrict__ Qb, int q_ld, int q_off,
                          const u16* __restrict__ Kb, int k_ld, int k_off, int Tk,
                          const u16* __restrict__ Vb, int v_ld, int v_off,
                          u16* __restrict__ Ob,
                          int nheads, int causal, int nkeys) {
  __shared__ u16 v_lds[32 * 64];
  __shared__ u16 p_lds[16 * 32];
  const int lane = threadIdx.x & 31;
  const int g = lane >> 4, ln = lane & 15;
  const int bh = blockIdx.y;
  const int b = bh / nheads, h = bh % nheads;
  const int qbase = blockIdx.x * 16;
  const int Tq = 512;

  // Q fragments for dh-chunks [0,32) and [32,64); fold in 1/sqrt(64) = 0.125
  AFrag qa[2];
  {
    const u16* qrow = Qb + ((size_t)(b * Tq + qbase + ln)) * q_ld + q_off + h * 64;
#pragma unroll
    for (int c = 0; c < 2; ++c) {
#pragma unroll
      for (int i = 0; i < 8; ++i) {
        qa[c].u[i]     = f2bf(bf2f(qrow[c * 32 + 8 * g + i]) * 0.125f);
        qa[c].u[8 + i] = f2bf(bf2f(qrow[c * 32 + 16 + 8 * g + i]) * 0.125f);
      }
    }
  }

  float run_m[8], run_l[8];
  v8f zero = {0.f, 0.f, 0.f, 0.f, 0.f, 0.f, 0.f, 0.f};
  v8f oacc[4];
#pragma unroll
  for (int r = 0; r < 8; ++r) { run_m[r] = -1e30f; run_l[r] = 0.f; }
#pragma unroll
  for (int j = 0; j < 4; ++j) oacc[j] = zero;

  const u32 lds_vrow = (u32)(size_t)(v_lds + lane * 64);  // low 32 bits = LDS offset

  const int nkb = causal ? (qbase + 15) / 32 + 1 : (nkeys + 31) / 32;
  for (int kb = 0; kb < nkb; ++kb) {
    const int kbb = kb * 32;
    // --- stage V[kbb..kbb+31][0..63] into LDS asynchronously (one row/lane) ---
    {
      // prior-iteration DS reads of v_lds are on DScnt; async writes are on
      // ASYNCcnt and unordered with them -> drain DS first.
      asm volatile("s_wait_dscnt 0x0" ::: "memory");
      const u64 gsrc =
          (u64)(size_t)(Vb + ((size_t)(b * Tk + kbb + lane)) * v_ld + v_off + h * 64);
#pragma unroll
      for (int i = 0; i < 8; ++i) {
        asm volatile("global_load_async_to_lds_b128 %0, %1, off offset:%2"
                     :: "v"(lds_vrow), "v"(gsrc), "n"(i * 16) : "memory");
      }
    }
    // scores S[16q, 32k]: 2 key-tiles x 2 dh-chunks (overlaps the async copy)
    v8f sacc[2];
    sacc[0] = zero; sacc[1] = zero;
#pragma unroll
    for (int tile = 0; tile < 2; ++tile) {
      const u16* krow =
          Kb + ((size_t)(b * Tk + kbb + tile * 16 + ln)) * k_ld + k_off + h * 64 + 8 * g;
#pragma unroll
      for (int c = 0; c < 2; ++c) {
        AFrag kf;
        kf.q[0] = *(const uint4*)(krow + c * 32);
        kf.q[1] = *(const uint4*)(krow + c * 32 + 16);
        sacc[tile] = __builtin_amdgcn_wmma_f32_16x16x32_bf16(
            false, qa[c].v, false, kf.v, (short)0, sacc[tile], false, false);
      }
    }
    // online softmax over this 32-key block
    float p0v[8], p1v[8];
#pragma unroll
    for (int r = 0; r < 8; ++r) {
      const int row = qbase + r + 8 * g;
      float s0 = sacc[0][r], s1 = sacc[1][r];
      const int c0 = kbb + ln, c1 = kbb + 16 + ln;
      if (causal) {
        if (c0 > row) s0 = -1e30f;
        if (c1 > row) s1 = -1e30f;
      }
      float bm = fmaxf(s0, s1);
#pragma unroll
      for (int m = 1; m < 16; m <<= 1) bm = fmaxf(bm, __shfl_xor(bm, m, 32));
      const float nm = fmaxf(run_m[r], bm);
      const float sc = __expf(run_m[r] - nm);
      const float p0 = __expf(s0 - nm), p1 = __expf(s1 - nm);
      float rl = p0 + p1;
#pragma unroll
      for (int m = 1; m < 16; m <<= 1) rl += __shfl_xor(rl, m, 32);
      run_l[r] = run_l[r] * sc + rl;
      run_m[r] = nm;
#pragma unroll
      for (int j = 0; j < 4; ++j) oacc[j][r] *= sc;
      p0v[r] = p0; p1v[r] = p1;
    }
    __syncthreads();  // single-wave WG: lowers to S_NOP; LDS in-order per wave
#pragma unroll
    for (int r = 0; r < 8; ++r) {
      p_lds[(r + 8 * g) * 32 + ln]      = f2bf(p0v[r]);
      p_lds[(r + 8 * g) * 32 + 16 + ln] = f2bf(p1v[r]);
    }
    __syncthreads();
    asm volatile("s_wait_asynccnt 0x0" ::: "memory");  // V tile landed in LDS
    // P in A-layout from LDS; V in B-layout from LDS; accumulate O += P @ V
    AFrag pa;
#pragma unroll
    for (int i = 0; i < 8; ++i) {
      pa.u[i]     = p_lds[ln * 32 + 8 * g + i];
      pa.u[8 + i] = p_lds[ln * 32 + 16 + 8 * g + i];
    }
#pragma unroll
    for (int j = 0; j < 4; ++j) {
      AFrag vf;
#pragma unroll
      for (int i = 0; i < 16; ++i) vf.u[i] = v_lds[(i < 8 ? 8 * g + i : 8 + 8 * g + i) * 64 + j * 16 + ln];
      oacc[j] = __builtin_amdgcn_wmma_f32_16x16x32_bf16(
          false, pa.v, false, vf.v, (short)0, oacc[j], false, false);
    }
  }

  // normalize + write merged-head output [b, t, h*64 + dh] as bf16
#pragma unroll
  for (int r = 0; r < 8; ++r) {
    const int row = qbase + r + 8 * g;
    const float inv = 1.f / run_l[r];
    u16* orow = Ob + ((size_t)(b * Tq + row)) * 256 + h * 64;
#pragma unroll
    for (int j = 0; j < 4; ++j) orow[j * 16 + ln] = f2bf(oacc[j][r] * inv);
  }
}

// ---------------------------------------------------------------------------
// Kernel 8: residual + LayerNorm over D=256 (one wave per row)
// ---------------------------------------------------------------------------
__global__ void residual_ln(const float* __restrict__ X, const float* __restrict__ Y,
                            const float* __restrict__ gamma, const float* __restrict__ beta,
                            float* __restrict__ Of, u16* __restrict__ Ob) {
  const int row = blockIdx.x;
  const int lane = threadIdx.x;
  float v[8];
  float s = 0.f, s2 = 0.f;
#pragma unroll
  for (int i = 0; i < 8; ++i) {
    const float t = X[(size_t)row * 256 + i * 32 + lane] + Y[(size_t)row * 256 + i * 32 + lane];
    v[i] = t; s += t; s2 += t * t;
  }
#pragma unroll
  for (int m = 1; m < 32; m <<= 1) {
    s  += __shfl_xor(s, m, 32);
    s2 += __shfl_xor(s2, m, 32);
  }
  const float mean = s * (1.f / 256.f);
  const float var  = s2 * (1.f / 256.f) - mean * mean;
  const float rstd = rsqrtf(var + 1e-5f);
#pragma unroll
  for (int i = 0; i < 8; ++i) {
    const int d = i * 32 + lane;
    const float o = (v[i] - mean) * rstd * gamma[d] + beta[d];
    if (Of) Of[(size_t)row * 256 + d] = o;
    Ob[(size_t)row * 256 + d] = f2bf(o);
  }
}

// ---------------------------------------------------------------------------
// Host side
// ---------------------------------------------------------------------------
extern "C" void kernel_launch(void* const* d_in, const int* in_sizes, int n_in,
                              void* d_out, int out_size, void* d_ws, size_t ws_size,
                              hipStream_t stream) {
  (void)in_sizes; (void)n_in; (void)out_size; (void)ws_size;
  const float* enc    = (const float*)d_in[0];
  const float* caps   = (const float*)d_in[1];
  const int*   cl     = (const int*)  d_in[2];
  const float* w_enc  = (const float*)d_in[3];
  const float* b_enc  = (const float*)d_in[4];
  const float* w_emb  = (const float*)d_in[5];
  const float* b_emb  = (const float*)d_in[6];
  const float* sa_w   = (const float*)d_in[7];
  const float* sa_b   = (const float*)d_in[8];
  const float* sa_ow  = (const float*)d_in[9];
  const float* sa_ob  = (const float*)d_in[10];
  const float* ca_w   = (const float*)d_in[11];
  const float* ca_b   = (const float*)d_in[12];
  const float* ca_ow  = (const float*)d_in[13];
  const float* ca_ob  = (const float*)d_in[14];
  const float* l1_w   = (const float*)d_in[15];
  const float* l1_b   = (const float*)d_in[16];
  const float* l2_w   = (const float*)d_in[17];
  const float* l2_b   = (const float*)d_in[18];
  const float* n1_g   = (const float*)d_in[19];
  const float* n1_b   = (const float*)d_in[20];
  const float* n2_g   = (const float*)d_in[21];
  const float* n2_b   = (const float*)d_in[22];
  const float* n3_g   = (const float*)d_in[23];
  const float* n3_b   = (const float*)d_in[24];
  const float* out_w  = (const float*)d_in[25];
  const float* out_b  = (const float*)d_in[26];
  float* out = (float*)d_out;

  // ---- workspace layout (256B aligned carves) ----
  char* ws = (char*)d_ws;
  size_t off = 0;
  auto alloc = [&](size_t bytes) -> void* {
    void* p = ws + off;
    off = (off + bytes + 255) & ~(size_t)255;
    return p;
  };
  int*  order   = (int*)alloc(32 * 4);
  int*  declen  = (int*)alloc(32 * 4);
  u16*  big     = (u16*)alloc((size_t)16384 * 2048 * 2);  // enc_bf early, ff1_bf later
  u16*  wenc_bf = (u16*)alloc((size_t)256 * 2048 * 2);
  u16*  saw_bf  = (u16*)alloc((size_t)768 * 256 * 2);
  u16*  saow_bf = (u16*)alloc((size_t)256 * 256 * 2);
  u16*  caw_bf  = (u16*)alloc((size_t)768 * 256 * 2);
  u16*  caow_bf = (u16*)alloc((size_t)256 * 256 * 2);
  u16*  l1_bf   = (u16*)alloc((size_t)2048 * 256 * 2);
  u16*  l2_bf   = (u16*)alloc((size_t)256 * 2048 * 2);
  u16*  ow_bf   = (u16*)alloc((size_t)8000 * 256 * 2);
  u16*  mem_bf  = (u16*)alloc((size_t)8192 * 256 * 2);
  u16*  x_bf    = (u16*)alloc((size_t)16384 * 256 * 2);
  u16*  qkv_bf  = (u16*)alloc((size_t)16384 * 768 * 2);   // later reused as qc|kc|vc
  u16*  ao_bf   = (u16*)alloc((size_t)16384 * 256 * 2);
  float* proj_f = (float*)alloc((size_t)16384 * 256 * 4);
  float* xA_f   = (float*)alloc((size_t)16384 * 256 * 4);
  float* xB_f   = (float*)alloc((size_t)16384 * 256 * 4);
  u16* qc_bf = qkv_bf;                                    // [16384,256]
  u16* kc_bf = qkv_bf + (size_t)16384 * 256;              // [8192,256]
  u16* vc_bf = qkv_bf + (size_t)16384 * 256 + (size_t)8192 * 256;

  // ---- 1. sort + tuple tail outputs ----
  hipLaunchKernelGGL(prep_kernel, dim3(1), dim3(32), 0, stream, cl, order, declen);
  hipLaunchKernelGGL(finalize_kernel, dim3((16448 + 255) / 256), dim3(256), 0, stream,
                     caps, order, declen, out);

  // ---- 2. weight conversion to bf16 ----
  auto cvt = [&](const float* s, u16* d, int n) {
    hipLaunchKernelGGL(cvt_bf16, dim3((n + 255) / 256), dim3(256), 0, stream, s, d, n);
  };
  cvt(w_enc, wenc_bf, 256 * 2048);
  cvt(sa_w,  saw_bf,  768 * 256);
  cvt(sa_ow, saow_bf, 256 * 256);
  cvt(ca_w,  caw_bf,  768 * 256);
  cvt(ca_ow, caow_bf, 256 * 256);
  cvt(l1_w,  l1_bf,   2048 * 256);
  cvt(l2_w,  l2_bf,   256 * 2048);
  cvt(out_w, ow_bf,   8000 * 256);

  // ---- 3. memory path: gather enc by order, mem = enc@w_enc^T + b_enc + PE ----
  hipLaunchKernelGGL(gather_enc, dim3(2048), dim3(256), 0, stream, enc, order, big);
  hipLaunchKernelGGL(gemm_wmma, dim3(256 / 64, 8192 / 128), dim3(128), 0, stream,
                     big, wenc_bf, b_enc, (float*)nullptr, mem_bf,
                     8192, 256, 2048, 1, (const int*)nullptr);

  // ---- 4. target embedding -> x0 ----
  hipLaunchKernelGGL(embed_kernel, dim3(16384 * 256 / 256), dim3(256), 0, stream,
                     caps, order, w_emb, b_emb, xA_f, x_bf);

  // ---- 5. self-attention ----
  hipLaunchKernelGGL(gemm_wmma, dim3(768 / 64, 16384 / 128), dim3(128), 0, stream,
                     x_bf, saw_bf, sa_b, (float*)nullptr, qkv_bf,
                     16384, 768, 256, 0, (const int*)nullptr);
  hipLaunchKernelGGL(attn_wmma, dim3(32, 128), dim3(32), 0, stream,
                     qkv_bf, 768, 0, qkv_bf, 768, 256, 512, qkv_bf, 768, 512,
                     ao_bf, 4, 1, 512);
  hipLaunchKernelGGL(gemm_wmma, dim3(256 / 64, 16384 / 128), dim3(128), 0, stream,
                     ao_bf, saow_bf, sa_ob, proj_f, (u16*)nullptr,
                     16384, 256, 256, 3, (const int*)nullptr);
  hipLaunchKernelGGL(residual_ln, dim3(16384), dim3(32), 0, stream,
                     xA_f, proj_f, n1_g, n1_b, xB_f, x_bf);

  // ---- 6. cross-attention ----
  hipLaunchKernelGGL(gemm_wmma, dim3(256 / 64, 16384 / 128), dim3(128), 0, stream,
                     x_bf, caw_bf, ca_b, (float*)nullptr, qc_bf,
                     16384, 256, 256, 0, (const int*)nullptr);
  hipLaunchKernelGGL(gemm_wmma, dim3(256 / 64, 8192 / 128), dim3(128), 0, stream,
                     mem_bf, caw_bf + (size_t)256 * 256, ca_b + 256, (float*)nullptr, kc_bf,
                     8192, 256, 256, 0, (const int*)nullptr);
  hipLaunchKernelGGL(gemm_wmma, dim3(256 / 64, 8192 / 128), dim3(128), 0, stream,
                     mem_bf, caw_bf + (size_t)512 * 256, ca_b + 512, (float*)nullptr, vc_bf,
                     8192, 256, 256, 0, (const int*)nullptr);
  hipLaunchKernelGGL(attn_wmma, dim3(32, 128), dim3(32), 0, stream,
                     qc_bf, 256, 0, kc_bf, 256, 0, 256, vc_bf, 256, 0,
                     ao_bf, 4, 0, 256);
  hipLaunchKernelGGL(gemm_wmma, dim3(256 / 64, 16384 / 128), dim3(128), 0, stream,
                     ao_bf, caow_bf, ca_ob, proj_f, (u16*)nullptr,
                     16384, 256, 256, 3, (const int*)nullptr);
  hipLaunchKernelGGL(residual_ln, dim3(16384), dim3(32), 0, stream,
                     xB_f, proj_f, n2_g, n2_b, xA_f, x_bf);

  // ---- 7. feed-forward ----
  hipLaunchKernelGGL(gemm_wmma, dim3(2048 / 64, 16384 / 128), dim3(128), 0, stream,
                     x_bf, l1_bf, l1_b, (float*)nullptr, big,
                     16384, 2048, 256, 2, (const int*)nullptr);
  hipLaunchKernelGGL(gemm_wmma, dim3(256 / 64, 16384 / 128), dim3(128), 0, stream,
                     big, l2_bf, l2_b, proj_f, (u16*)nullptr,
                     16384, 256, 2048, 3, (const int*)nullptr);
  hipLaunchKernelGGL(residual_ln, dim3(16384), dim3(32), 0, stream,
                     xA_f, proj_f, n3_g, n3_b, (float*)nullptr, x_bf);

  // ---- 8. logits + validity mask straight into d_out ----
  hipLaunchKernelGGL(gemm_wmma, dim3(8000 / 64, 16384 / 128), dim3(128), 0, stream,
                     x_bf, ow_bf, out_b, out, (u16*)nullptr,
                     16384, 8000, 256, 4, declen);
}
